// DeepseekV2Attention_12764642804319
// MI455X (gfx1250) — matrix operand order repaired and checked by reading, not compiled, not run on hardware
//
#include <hip/hip_runtime.h>
#include <cmath>

// ---------------- problem constants ----------------
constexpr int S_LEN  = 2048;
constexpr int HIDDEN = 5120;
constexpr int NH     = 16;
constexpr int NOPE   = 128;
constexpr int ROPE_D = 64;
constexpr int VD     = 128;
constexpr int QK     = NOPE + ROPE_D;   // 192
constexpr int QLORA  = 1536;
constexpr int KVLORA = 512;
constexpr float EPS  = 1e-6f;

typedef __attribute__((ext_vector_type(16))) __bf16 bf16x16;
typedef __attribute__((ext_vector_type(8)))  __bf16 bf16x8;
typedef __attribute__((ext_vector_type(8)))  float  f32x8;

static __device__ __forceinline__ unsigned short f2bf(float f) {
  unsigned u = __builtin_bit_cast(unsigned, f);
  u += 0x7FFFu + ((u >> 16) & 1u);            // round-to-nearest-even
  return (unsigned short)(u >> 16);
}

// ---------------- elementwise converts ----------------
__global__ void k_f32_to_bf16(const float* __restrict__ in,
                              unsigned short* __restrict__ out, long n) {
  long i = (long)blockIdx.x * blockDim.x + threadIdx.x;
  if (i < n) out[i] = f2bf(in[i]);
}

// in: [K,N] row-major f32  ->  out: [N,K] row-major bf16   (B^T for TN gemm)
__global__ void k_transpose_bf16(const float* __restrict__ in,
                                 unsigned short* __restrict__ out, int K, int N) {
  long i = (long)blockIdx.x * blockDim.x + threadIdx.x;
  if (i < (long)K * N) {
    int n = (int)(i / K);
    int k = (int)(i % K);
    out[i] = f2bf(in[(long)k * N + n]);
  }
}

// ---------------- TN bf16 WMMA GEMM ----------------
// C[M,N] (f32, leading dim ldc) = A[M,K] (bf16 row-major) * Bt[N,K]^T (bf16 row-major)
// One wave -> 32x64 C tile: 2 M-subtiles x 4 N-subtiles = 8 WMMA accumulators.
// Per 32-K step: 4 A + 8 B b128 loads (hoisted base pointers, immediate offsets)
// feed 8 back-to-back WMMAs; loads for the next step can overlap the WMMA block.
// Operand layouts (A/B K order is this kernel's contiguous choice, not the ISA table order):
//   A 16x32 bf16 : per-lane 16 contiguous K (group 0/1 -> K+0/K+16), M = lane&15
//   B 32x16 bf16 : per-lane 16 contiguous K (group 0/1 -> K+0/K+16), N = lane&15
//   C 16x16 f32  : elem i -> row (grp?8:0)+i, col = lane&15
__global__ void __launch_bounds__(256, 1)
k_gemm_tn_bf16(const unsigned short* __restrict__ A,
               const unsigned short* __restrict__ Bt,
               float* __restrict__ C,
               int M, int N, int Kd, int ldc, int causal) {
  const int wave = blockIdx.x * (blockDim.x >> 5) + (threadIdx.x >> 5);
  const int lane = threadIdx.x & 31;
  const int ntiles = N >> 6;                        // 64-wide tiles
  const int tm = wave / ntiles;
  const int tn = wave % ntiles;
  if (tm >= (M >> 5)) return;                       // whole-wave uniform
  if (causal && (tn * 64) > (tm * 32 + 31)) return; // skip fully-masked tiles

  const int lh  = lane & 15;
  const int grp = lane >> 4;                        // 0 or 1
  const unsigned short* pa0 = A + (size_t)(tm * 32 + lh) * Kd + grp * 16;
  const unsigned short* pa1 = pa0 + (size_t)16 * Kd;
  const unsigned short* pb0 = Bt + (size_t)(tn * 64 + lh) * Kd + grp * 16;
  const unsigned short* pb1 = pb0 + (size_t)16 * Kd;
  const unsigned short* pb2 = pb0 + (size_t)32 * Kd;
  const unsigned short* pb3 = pb0 + (size_t)48 * Kd;

  f32x8 acc0 = {}, acc1 = {}, acc2 = {}, acc3 = {};
  f32x8 acc4 = {}, acc5 = {}, acc6 = {}, acc7 = {};

  for (int kk = 0; kk < Kd; kk += 32) {
    // streaming prefetch a few K-steps ahead (global_prefetch_b8)
    __builtin_prefetch(pa0 + kk + 256, 0, 1);
    __builtin_prefetch(pb0 + kk + 256, 0, 1);

    // ---- all operand loads for this K-step first (clause-friendly) ----
    bf16x8 a0lo = *(const bf16x8*)(pa0 + kk);
    bf16x8 a0hi = *(const bf16x8*)(pa0 + kk + 8);
    bf16x8 a1lo = *(const bf16x8*)(pa1 + kk);
    bf16x8 a1hi = *(const bf16x8*)(pa1 + kk + 8);
    bf16x16 b0 = *(const bf16x16*)(pb0 + kk);
    bf16x16 b1 = *(const bf16x16*)(pb1 + kk);
    bf16x16 b2 = *(const bf16x16*)(pb2 + kk);
    bf16x16 b3 = *(const bf16x16*)(pb3 + kk);
    bf16x16 a0 = __builtin_shufflevector(a0lo, a0hi,
        0,1,2,3,4,5,6,7,8,9,10,11,12,13,14,15);
    bf16x16 a1 = __builtin_shufflevector(a1lo, a1hi,
        0,1,2,3,4,5,6,7,8,9,10,11,12,13,14,15);

    // ---- 8 back-to-back WMMAs ----
    acc0 = __builtin_amdgcn_wmma_f32_16x16x32_bf16(false, a0, false, b0, (short)0, acc0, false, false);
    acc4 = __builtin_amdgcn_wmma_f32_16x16x32_bf16(false, a1, false, b0, (short)0, acc4, false, false);
    acc1 = __builtin_amdgcn_wmma_f32_16x16x32_bf16(false, a0, false, b1, (short)0, acc1, false, false);
    acc5 = __builtin_amdgcn_wmma_f32_16x16x32_bf16(false, a1, false, b1, (short)0, acc5, false, false);
    acc2 = __builtin_amdgcn_wmma_f32_16x16x32_bf16(false, a0, false, b2, (short)0, acc2, false, false);
    acc6 = __builtin_amdgcn_wmma_f32_16x16x32_bf16(false, a1, false, b2, (short)0, acc6, false, false);
    acc3 = __builtin_amdgcn_wmma_f32_16x16x32_bf16(false, a0, false, b3, (short)0, acc3, false, false);
    acc7 = __builtin_amdgcn_wmma_f32_16x16x32_bf16(false, a1, false, b3, (short)0, acc7, false, false);
  }

  f32x8 accs[8] = {acc0, acc1, acc2, acc3, acc4, acc5, acc6, acc7};
#pragma unroll
  for (int sm = 0; sm < 2; ++sm) {
    const int rowb = tm * 32 + sm * 16 + grp * 8;
#pragma unroll
    for (int sub = 0; sub < 4; ++sub) {
      const int col = tn * 64 + sub * 16 + lh;
#pragma unroll
      for (int i = 0; i < 8; ++i)
        C[(size_t)(rowb + i) * ldc + col] = accs[sm * 4 + sub][i];
    }
  }
}

// ---------------- RMSNorm (f32 in, bf16 out) ----------------
__global__ void k_rmsnorm_bf16(const float* __restrict__ in, int ld_in,
                               const float* __restrict__ w,
                               unsigned short* __restrict__ out, int D) {
  const int r = blockIdx.x;
  const float* x = in + (size_t)r * ld_in;
  __shared__ float red[256];
  float s = 0.f;
  for (int i = threadIdx.x; i < D; i += 256) { float v = x[i]; s += v * v; }
  red[threadIdx.x] = s; __syncthreads();
  for (int st = 128; st > 0; st >>= 1) {
    if (threadIdx.x < st) red[threadIdx.x] += red[threadIdx.x + st];
    __syncthreads();
  }
  const float rs = rsqrtf(red[0] / (float)D + EPS);
  for (int i = threadIdx.x; i < D; i += 256)
    out[(size_t)r * D + i] = f2bf(x[i] * rs * w[i]);
}

// ---------------- YaRN RoPE helpers ----------------
static __device__ __forceinline__ float yarn_inv_freq(int i, float rlow, float rhigh) {
  float pf = powf(10000.0f, (2.0f * (float)i) / 64.0f);
  float ramp = ((float)i - rlow) / fmaxf(rhigh - rlow, 0.001f);
  ramp = fminf(fmaxf(ramp, 0.0f), 1.0f);
  return (1.0f / (40.0f * pf)) * ramp + (1.0f / pf) * (1.0f - ramp);
}

// qf[h][s][0..191] : nope copy + roped q_pe
__global__ void k_build_qf(const float* __restrict__ q, const int* __restrict__ pos,
                           unsigned short* __restrict__ qf, float rlow, float rhigh) {
  const int b = blockIdx.x;            // b = h*S + s
  const int h = b / S_LEN, s = b % S_LEN;
  const int d = threadIdx.x;           // 0..191
  const float* src = q + (size_t)s * (NH * QK) + h * QK;
  float val;
  if (d < NOPE) {
    val = src[d];
  } else {
    const int j = d - NOPE, i = j >> 1;
    const float x1 = src[NOPE + 2 * i], x2 = src[NOPE + 2 * i + 1];
    const float fr = (float)pos[s] * yarn_inv_freq(i, rlow, rhigh);
    const float c = cosf(fr), sn = sinf(fr);
    val = (j & 1) ? (x2 * c + x1 * sn) : (x1 * c - x2 * sn);
  }
  qf[((size_t)h * S_LEN + s) * QK + d] = f2bf(val);
}

// kf[h][s][0..191] : k_nope copy + roped shared k_pe (broadcast over heads)
__global__ void k_build_kf(const float* __restrict__ kv, const float* __restrict__ latent,
                           const int* __restrict__ pos,
                           unsigned short* __restrict__ kf, float rlow, float rhigh) {
  const int b = blockIdx.x;
  const int h = b / S_LEN, s = b % S_LEN;
  const int d = threadIdx.x;
  float val;
  if (d < NOPE) {
    val = kv[(size_t)s * (NH * (NOPE + VD)) + h * (NOPE + VD) + d];
  } else {
    const int j = d - NOPE, i = j >> 1;
    const float* pe = latent + (size_t)s * (KVLORA + ROPE_D) + KVLORA;
    const float x1 = pe[2 * i], x2 = pe[2 * i + 1];
    const float fr = (float)pos[s] * yarn_inv_freq(i, rlow, rhigh);
    const float c = cosf(fr), sn = sinf(fr);
    val = (j & 1) ? (x2 * c + x1 * sn) : (x1 * c - x2 * sn);
  }
  kf[((size_t)h * S_LEN + s) * QK + d] = f2bf(val);
}

// vT[h][d][s] = v part of kv  (pre-transposed B^T for the PV gemm)
__global__ void k_build_vT(const float* __restrict__ kv, unsigned short* __restrict__ vT) {
  long i = (long)blockIdx.x * blockDim.x + threadIdx.x;
  if (i >= (long)NH * VD * S_LEN) return;
  const int s = (int)(i % S_LEN);
  const int d = (int)((i / S_LEN) % VD);
  const int h = (int)(i / ((long)VD * S_LEN));
  vT[i] = f2bf(kv[(size_t)s * (NH * (NOPE + VD)) + h * (NOPE + VD) + NOPE + d]);
}

// causal softmax over one score row; scale fused; writes bf16 probs (0 past diag)
__global__ void k_softmax_causal(const float* __restrict__ scores,
                                 unsigned short* __restrict__ probs, float scaling) {
  const int q = blockIdx.x;
  const float* row = scores + (size_t)q * S_LEN;
  __shared__ float red[256];
  const int t = threadIdx.x;
  float m = -3.0e38f;
  for (int k = t; k <= q; k += 256) m = fmaxf(m, row[k] * scaling);
  red[t] = m; __syncthreads();
  for (int st = 128; st > 0; st >>= 1) {
    if (t < st) red[t] = fmaxf(red[t], red[t + st]);
    __syncthreads();
  }
  m = red[0]; __syncthreads();
  float sum = 0.f;
  for (int k = t; k <= q; k += 256) sum += expf(row[k] * scaling - m);
  red[t] = sum; __syncthreads();
  for (int st = 128; st > 0; st >>= 1) {
    if (t < st) red[t] += red[t + st];
    __syncthreads();
  }
  const float inv = 1.0f / red[0];
  unsigned short* prow = probs + (size_t)q * S_LEN;
  for (int k = t; k < S_LEN; k += 256)
    prow[k] = (k <= q) ? f2bf(expf(row[k] * scaling - m) * inv) : (unsigned short)0;
}

// ---------------- host side ----------------
extern "C" void kernel_launch(void* const* d_in, const int* in_sizes, int n_in,
                              void* d_out, int out_size, void* d_ws, size_t ws_size,
                              hipStream_t stream) {
  (void)in_sizes; (void)n_in; (void)out_size; (void)ws_size;
  const int*   positions = (const int*)d_in[0];
  const float* hidden    = (const float*)d_in[1];
  const float* w_q_a     = (const float*)d_in[2];
  const float* q_a_ln_w  = (const float*)d_in[3];
  const float* w_q_b     = (const float*)d_in[4];
  const float* w_kv_a    = (const float*)d_in[5];
  const float* kv_a_ln_w = (const float*)d_in[6];
  const float* w_kv_b    = (const float*)d_in[7];
  const float* w_o       = (const float*)d_in[8];
  float* out = (float*)d_out;

  // YaRN constants (host-side)
  const double two_pi = 6.283185307179586;
  auto corr_dim = [&](double n_rot) {
    return ROPE_D * std::log(4096.0 / (n_rot * two_pi)) / (2.0 * std::log(10000.0));
  };
  const float rlow  = (float)std::fmax(std::floor(corr_dim(32.0)), 0.0);
  const float rhigh = (float)std::fmin(std::ceil(corr_dim(1.0)), (double)(ROPE_D - 1));
  const double msc = 0.1 * 0.707 * std::log(40.0) + 1.0;
  const float scaling = (float)(std::pow((double)QK, -0.5) * msc * msc);

  // workspace carve-up
  size_t off = 0;
  auto carve = [&](size_t bytes) {
    char* p = (char*)d_ws + off;
    off += (bytes + 255) & ~(size_t)255;
    return p;
  };
  unsigned short* hs_b   = (unsigned short*)carve((size_t)S_LEN * HIDDEN * 2);
  unsigned short* wqaT   = (unsigned short*)carve((size_t)QLORA * HIDDEN * 2);
  float*          qa_f   = (float*)carve((size_t)S_LEN * QLORA * 4);
  unsigned short* qa_b   = (unsigned short*)carve((size_t)S_LEN * QLORA * 2);
  unsigned short* wqbT   = (unsigned short*)carve((size_t)NH * QK * QLORA * 2);
  float*          q_f    = (float*)carve((size_t)S_LEN * NH * QK * 4);
  unsigned short* wkvaT  = (unsigned short*)carve((size_t)(KVLORA + ROPE_D) * HIDDEN * 2);
  float*          lat_f  = (float*)carve((size_t)S_LEN * (KVLORA + ROPE_D) * 4);
  unsigned short* kva_b  = (unsigned short*)carve((size_t)S_LEN * KVLORA * 2);
  unsigned short* wkvbT  = (unsigned short*)carve((size_t)NH * (NOPE + VD) * KVLORA * 2);
  float*          kv_f   = (float*)carve((size_t)S_LEN * NH * (NOPE + VD) * 4);
  unsigned short* qf_b   = (unsigned short*)carve((size_t)NH * S_LEN * QK * 2);
  unsigned short* kf_b   = (unsigned short*)carve((size_t)NH * S_LEN * QK * 2);
  unsigned short* vT_b   = (unsigned short*)carve((size_t)NH * VD * S_LEN * 2);
  float*          sc_f   = (float*)carve((size_t)S_LEN * S_LEN * 4);
  unsigned short* pr_b   = (unsigned short*)carve((size_t)S_LEN * S_LEN * 2);
  float*          attn_f = (float*)carve((size_t)S_LEN * NH * VD * 4);
  unsigned short* attn_b = (unsigned short*)carve((size_t)S_LEN * NH * VD * 2);
  unsigned short* woT    = (unsigned short*)carve((size_t)HIDDEN * NH * VD * 2);

  auto cvt = [&](const float* in, unsigned short* o, long n) {
    k_f32_to_bf16<<<(unsigned)((n + 255) / 256), 256, 0, stream>>>(in, o, n);
  };
  auto trp = [&](const float* in, unsigned short* o, int K, int N) {
    long n = (long)K * N;
    k_transpose_bf16<<<(unsigned)((n + 255) / 256), 256, 0, stream>>>(in, o, K, N);
  };
  auto gemm = [&](const unsigned short* A, const unsigned short* Bt, float* C,
                  int M, int N, int Kd, int ldc, int causal) {
    int tiles = (M / 32) * (N / 64);
    k_gemm_tn_bf16<<<(tiles + 7) / 8, 256, 0, stream>>>(A, Bt, C, M, N, Kd, ldc, causal);
  };

  // ---- projections ----
  cvt(hidden, hs_b, (long)S_LEN * HIDDEN);
  trp(w_q_a, wqaT, HIDDEN, QLORA);
  gemm(hs_b, wqaT, qa_f, S_LEN, QLORA, HIDDEN, QLORA, 0);
  k_rmsnorm_bf16<<<S_LEN, 256, 0, stream>>>(qa_f, QLORA, q_a_ln_w, qa_b, QLORA);
  trp(w_q_b, wqbT, QLORA, NH * QK);
  gemm(qa_b, wqbT, q_f, S_LEN, NH * QK, QLORA, NH * QK, 0);

  trp(w_kv_a, wkvaT, HIDDEN, KVLORA + ROPE_D);
  gemm(hs_b, wkvaT, lat_f, S_LEN, KVLORA + ROPE_D, HIDDEN, KVLORA + ROPE_D, 0);
  k_rmsnorm_bf16<<<S_LEN, 256, 0, stream>>>(lat_f, KVLORA + ROPE_D, kv_a_ln_w, kva_b, KVLORA);
  trp(w_kv_b, wkvbT, KVLORA, NH * (NOPE + VD));
  gemm(kva_b, wkvbT, kv_f, S_LEN, NH * (NOPE + VD), KVLORA, NH * (NOPE + VD), 0);

  // ---- RoPE + head-major operand assembly ----
  k_build_qf<<<NH * S_LEN, QK, 0, stream>>>(q_f, positions, qf_b, rlow, rhigh);
  k_build_kf<<<NH * S_LEN, QK, 0, stream>>>(kv_f, lat_f, positions, kf_b, rlow, rhigh);
  {
    long n = (long)NH * VD * S_LEN;
    k_build_vT<<<(unsigned)((n + 255) / 256), 256, 0, stream>>>(kv_f, vT_b);
  }

  // ---- per-head attention (serialized on stream; scratch reused) ----
  for (int h = 0; h < NH; ++h) {
    const unsigned short* qf_h = qf_b + (size_t)h * S_LEN * QK;
    const unsigned short* kf_h = kf_b + (size_t)h * S_LEN * QK;
    const unsigned short* vT_h = vT_b + (size_t)h * VD * S_LEN;
    gemm(qf_h, kf_h, sc_f, S_LEN, S_LEN, QK, S_LEN, /*causal=*/1);
    k_softmax_causal<<<S_LEN, 256, 0, stream>>>(sc_f, pr_b, scaling);
    gemm(pr_b, vT_h, attn_f + h * VD, S_LEN, VD, S_LEN, NH * VD, 0);
  }

  // ---- output projection ----
  cvt(attn_f, attn_b, (long)S_LEN * NH * VD);
  trp(w_o, woT, NH * VD, HIDDEN);
  gemm(attn_b, woT, out, S_LEN, HIDDEN, NH * VD, HIDDEN, 0);
}
